// BertSelfAttention_26980984554199
// MI455X (gfx1250) — compile-verified, matrix operand order checked by reading, not run on hardware
//
#include <hip/hip_runtime.h>
#include <hip/hip_bf16.h>

// ---------------------------------------------------------------------------
// BERT self-attention for MI455X (gfx1250, wave32, WMMA 16x16x32 f16,
// async global->LDS staging double-buffered on ASYNCcnt).
// B=4, S=2048, H=1024, NH=16, HD=64.
// Pipeline:
//   1) cvt_f32_f16 : X, Wq, Wk, Wv  -> f16 workspace
//   2) qkv_gemm x3 : Q/K/V = X@W + b, stored f16 in [B,NH,S,HD]
//   3) attention   : flash-style online-softmax attention, f32 output
// ---------------------------------------------------------------------------

typedef __attribute__((ext_vector_type(16))) _Float16 v16h;
typedef __attribute__((ext_vector_type(8)))  _Float16 v8h;
typedef __attribute__((ext_vector_type(4)))  _Float16 v4h;
typedef __attribute__((ext_vector_type(8)))  float    v8f;
typedef __attribute__((ext_vector_type(4)))  float    v4f;

#define LOG2E 1.4426950408889634f

__device__ __forceinline__ v8f wmma_f16(v16h a, v16h b, v8f c) {
  // D = A(16x32 f16) x B(32x16 f16) + C(16x16 f32)
  return __builtin_amdgcn_wmma_f32_16x16x32_f16(
      /*neg_a=*/false, a, /*neg_b=*/false, b,
      /*c_mod=*/(short)0, c, /*reuse_a=*/false, /*reuse_b=*/false);
}

// 16x32 f16 A-fragment per ISA 7.12.2 (wave32): lane L holds row L%16 with
// K = {kb..kb+7, kb+16..kb+23}, kb = (L/16)*8.  Two 16B loads per lane.
// Also used for B fragments from column-contiguous staging (lane = column).
__device__ __forceinline__ v16h frag_ld(const _Float16* base, int row, int stride) {
  const int lane = threadIdx.x & 31;
  const _Float16* p = base + (row + (lane & 15)) * stride + ((lane >> 4) << 3);
  v8h lo = *(const v8h*)(p);
  v8h hi = *(const v8h*)(p + 16);
  v16h f;
#pragma unroll
  for (int i = 0; i < 8; ++i) { f[i] = lo[i]; f[i + 8] = hi[i]; }
  return f;
}

// Async DMA of 16 bytes: global -> LDS, tracked on ASYNCcnt (no VGPR data).
// LDS address = low 32 bits of the generic pointer (ISA flat aperture rule).
__device__ __forceinline__ void async_cp16(const _Float16* gsrc, _Float16* ldst) {
  const unsigned long long ga = (unsigned long long)(uintptr_t)gsrc;
  const unsigned int       la = (unsigned int)(uintptr_t)ldst;
  asm volatile("global_load_async_to_lds_b128 %0, %1, off"
               :: "v"(la), "v"(ga) : "memory");
}
__device__ __forceinline__ void async_wait0() {
  asm volatile("s_wait_asynccnt 0" ::: "memory");
}

// ---------------------------------------------------------------------------
// Kernel 0: f32 -> f16 conversion (vectorized x4)
// ---------------------------------------------------------------------------
__global__ __launch_bounds__(256)
void cvt_f32_f16(const float* __restrict__ src, _Float16* __restrict__ dst, int n4) {
  const int i = blockIdx.x * 256 + threadIdx.x;
  if (i >= n4) return;
  v4f v = ((const v4f*)src)[i];
  v4h o;
#pragma unroll
  for (int k = 0; k < 4; ++k) o[k] = (_Float16)v[k];
  ((v4h*)dst)[i] = o;
}

// ---------------------------------------------------------------------------
// Kernel 1: Q/K/V projection GEMM.
//   X[8192,1024] f16 @ W[1024,1024] f16 + bias f32 -> Out f16 [B,NH,S,HD]
//   grid = (M/128, N/64); block = 256 (8 waves); N-tile(64) == one head.
//   K-step 64; X tile async double-buffered; W panel register-prefetched.
// ---------------------------------------------------------------------------
__global__ __launch_bounds__(256)
void qkv_gemm(const _Float16* __restrict__ X, const _Float16* __restrict__ W,
              const float* __restrict__ bias, _Float16* __restrict__ Out) {
  __shared__ _Float16 sX[2][128 * 72];  // 128 rows x 64 k (stride 72 keeps 16B align)
  __shared__ _Float16 sW[64 * 72];      // sW[n][k] : W^T panel, 64 n x 64 k

  const int tid  = threadIdx.x;
  const int wave = tid >> 5;
  const int lane = tid & 31;
  const int m0   = blockIdx.x * 128;
  const int n0   = blockIdx.y * 64;

  v8f acc[4] = {};

  // Prologue: async-stage X tile 0; sync-load W panel 0 into registers.
  {
#pragma unroll
    for (int c = 0; c < 4; ++c) {
      const int idx = tid + 256 * c;                 // 0..1023 16B chunks
      const int row = idx >> 3, ch = (idx & 7) << 3;
      async_cp16(X + (m0 + row) * 1024 + ch, sX[0] + row * 72 + ch);
    }
  }
  v8h wv[2];
#pragma unroll
  for (int c = 0; c < 2; ++c) {
    const int idx = tid + 256 * c;                   // 0..511
    const int k = idx >> 3, n8 = (idx & 7) << 3;
    wv[c] = *(const v8h*)(W + k * 1024 + n0 + n8);
  }

  int buf = 0;
  for (int kt = 0; kt < 1024; kt += 64, buf ^= 1) {
    async_wait0();
    __syncthreads();                                 // sX[buf] ready for all waves
    // Kick async stage of next X tile into the other buffer.
    if (kt + 64 < 1024) {
#pragma unroll
      for (int c = 0; c < 4; ++c) {
        const int idx = tid + 256 * c;
        const int row = idx >> 3, ch = (idx & 7) << 3;
        async_cp16(X + (m0 + row) * 1024 + (kt + 64) + ch, sX[buf ^ 1] + row * 72 + ch);
      }
    }
    // Scatter current W panel (registers) transposed into LDS; prefetch next.
#pragma unroll
    for (int c = 0; c < 2; ++c) {
      const int idx = tid + 256 * c;
      const int k = idx >> 3, n8 = (idx & 7) << 3;
#pragma unroll
      for (int i = 0; i < 8; ++i) sW[(n8 + i) * 72 + k] = wv[c][i];
      if (kt + 64 < 1024)
        wv[c] = *(const v8h*)(W + (kt + 64 + k) * 1024 + n0 + n8);
    }
    __syncthreads();                                 // sW visible

    const v16h a0 = frag_ld(sX[buf],      wave * 16, 72);   // k 0..31
    const v16h a1 = frag_ld(sX[buf] + 32, wave * 16, 72);   // k 32..63
#pragma unroll
    for (int j = 0; j < 4; ++j) {
      const v16h b0 = frag_ld(sW,      j * 16, 72);
      const v16h b1 = frag_ld(sW + 32, j * 16, 72);
      acc[j] = wmma_f16(a0, b0, acc[j]);
      acc[j] = wmma_f16(a1, b1, acc[j]);
    }
  }

  // Epilogue: bias add + f16 store into [B,NH,S,HD].
  const int r15  = lane & 15;
  const int off8 = (lane >> 4) << 3;
  const int h    = n0 >> 6;
#pragma unroll
  for (int j = 0; j < 4; ++j) {
    const int d  = j * 16 + r15;
    const float bv = bias[n0 + d];
#pragma unroll
    for (int r = 0; r < 8; ++r) {
      const int m  = m0 + wave * 16 + r + off8;   // global row in [0,8192)
      const int bb = m >> 11;                     // batch
      const int s  = m & 2047;                    // sequence pos
      Out[((size_t)((bb * 16 + h) * 2048 + s) << 6) + d] = (_Float16)(acc[j][r] + bv);
    }
  }
}

// ---------------------------------------------------------------------------
// Kernel 2: flash-attention. grid = (S/128, NH, B); block = 256 (8 waves).
// K and row-major V tiles async double-buffered; V transposed LDS->LDS.
// ---------------------------------------------------------------------------
__global__ __launch_bounds__(256)
void attention(const _Float16* __restrict__ Q, const _Float16* __restrict__ K,
               const _Float16* __restrict__ V, float* __restrict__ Out) {
  __shared__ _Float16 sK[2][64 * 72];     // K rows (K^T columns contiguous)
  __shared__ _Float16 sVr[2][64 * 72];    // V rows (async landing zone)
  __shared__ _Float16 sVt[64 * 72];       // sVt[d][kv] : V transposed
  __shared__ _Float16 sP[8 * 16 * 72];    // per-wave P staging (C -> A layout)

  const int tid  = threadIdx.x;
  const int wave = tid >> 5;
  const int lane = tid & 31;
  const int r15  = lane & 15;
  const int off8 = (lane >> 4) << 3;

  const size_t bh = (size_t)blockIdx.z * 16 + blockIdx.y;
  const _Float16* Qh = Q + bh * (2048 * 64);
  const _Float16* Kh = K + bh * (2048 * 64);
  const _Float16* Vh = V + bh * (2048 * 64);

  const int q0 = blockIdx.x * 128 + wave * 16;
  // Q fragments stay in registers for the whole pass (A-layout, stride HD=64).
  const v16h qa0 = frag_ld(Qh, q0, 64);
  const v16h qa1 = frag_ld(Qh + 32, q0, 64);

  float mrun[8], lrun[8];
#pragma unroll
  for (int r = 0; r < 8; ++r) { mrun[r] = -3.0e38f; lrun[r] = 0.0f; }
  v8f ctx[4] = {};

  _Float16* myP = sP + wave * (16 * 72);

  // Prologue: async-stage KV tile 0.
#pragma unroll
  for (int c = 0; c < 2; ++c) {
    const int idx = tid + 256 * c;
    const int row = idx >> 3, ch = (idx & 7) << 3;
    async_cp16(Kh + row * 64 + ch, sK[0]  + row * 72 + ch);
    async_cp16(Vh + row * 64 + ch, sVr[0] + row * 72 + ch);
  }

  int buf = 0;
  for (int kt = 0; kt < 2048; kt += 64, buf ^= 1) {
    async_wait0();
    __syncthreads();                       // sK[buf], sVr[buf] ready for all
    if (kt + 64 < 2048) {
#pragma unroll
      for (int c = 0; c < 2; ++c) {
        const int idx = tid + 256 * c;
        const int row = idx >> 3, ch = (idx & 7) << 3;
        async_cp16(Kh + (kt + 64 + row) * 64 + ch, sK[buf ^ 1]  + row * 72 + ch);
        async_cp16(Vh + (kt + 64 + row) * 64 + ch, sVr[buf ^ 1] + row * 72 + ch);
      }
    }
    // Transpose V tile: sVr[buf] -> sVt (b128 reads, b16 scatter).
#pragma unroll
    for (int c = 0; c < 2; ++c) {
      const int idx = tid + 256 * c;
      const int row = idx >> 3, ch = (idx & 7) << 3;
      v8h vv = *(const v8h*)(sVr[buf] + row * 72 + ch);
#pragma unroll
      for (int i = 0; i < 8; ++i) sVt[(ch + i) * 72 + row] = vv[i];
    }
    __syncthreads();                       // sVt visible

    // Scores: S = Q @ K^T (16x64 per wave, 8 WMMAs).
    v8f sc[4];
#pragma unroll
    for (int j = 0; j < 4; ++j) {
      const v16h kb0 = frag_ld(sK[buf],      j * 16, 72);
      const v16h kb1 = frag_ld(sK[buf] + 32, j * 16, 72);
      v8f a = {};
      a = wmma_f16(qa0, kb0, a);
      a = wmma_f16(qa1, kb1, a);
      sc[j] = a;
    }
#pragma unroll
    for (int j = 0; j < 4; ++j)
#pragma unroll
      for (int r = 0; r < 8; ++r) sc[j][r] *= 0.125f;   // 1/sqrt(HD)

    // Online softmax. C-layout: VGPR r holds rows (r, r+8) across lane halves;
    // row reduce = xor-shuffle over 16-lane groups (masks 8,4,2,1).
    float corr[8];
#pragma unroll
    for (int r = 0; r < 8; ++r) {
      float t = fmaxf(fmaxf(sc[0][r], sc[1][r]), fmaxf(sc[2][r], sc[3][r]));
#pragma unroll
      for (int m = 8; m >= 1; m >>= 1) t = fmaxf(t, __shfl_xor(t, m, 32));
      const float mnew = fmaxf(mrun[r], t);
      corr[r] = exp2f((mrun[r] - mnew) * LOG2E);
      mrun[r] = mnew;
    }
    float rsum[8];
#pragma unroll
    for (int r = 0; r < 8; ++r) rsum[r] = 0.0f;
#pragma unroll
    for (int j = 0; j < 4; ++j)
#pragma unroll
      for (int r = 0; r < 8; ++r) {
        const float p = exp2f((sc[j][r] - mrun[r]) * LOG2E);
        sc[j][r] = p;
        rsum[r] += p;
      }
#pragma unroll
    for (int r = 0; r < 8; ++r) {
      float t = rsum[r];
#pragma unroll
      for (int m = 8; m >= 1; m >>= 1) t += __shfl_xor(t, m, 32);
      lrun[r] = lrun[r] * corr[r] + t;
#pragma unroll
      for (int j = 0; j < 4; ++j) ctx[j][r] *= corr[r];
    }

    // P: C-layout -> per-wave LDS strip -> A-layout fragments.
#pragma unroll
    for (int j = 0; j < 4; ++j)
#pragma unroll
      for (int r = 0; r < 8; ++r)
        myP[(r + off8) * 72 + j * 16 + r15] = (_Float16)sc[j][r];
    asm volatile("s_wait_dscnt 0" ::: "memory");   // intra-wave DS RAW ordering
    const v16h pa0 = frag_ld(myP, 0, 72);
    const v16h pa1 = frag_ld(myP + 32, 0, 72);

    // ctx += P @ V (8 WMMAs; sVt columns of V contiguous).
#pragma unroll
    for (int jd = 0; jd < 4; ++jd) {
      const v16h vb0 = frag_ld(sVt,      jd * 16, 72);
      const v16h vb1 = frag_ld(sVt + 32, jd * 16, 72);
      ctx[jd] = wmma_f16(pa0, vb0, ctx[jd]);
      ctx[jd] = wmma_f16(pa1, vb1, ctx[jd]);
    }
  }

  // Epilogue: ctx / l  -> Out f32 [B,S,H].
  const int bb = blockIdx.z, h = blockIdx.y;
#pragma unroll
  for (int jd = 0; jd < 4; ++jd)
#pragma unroll
    for (int r = 0; r < 8; ++r) {
      const int s = q0 + r + off8;
      Out[(size_t)(bb * 2048 + s) * 1024 + h * 64 + jd * 16 + r15] =
          ctx[jd][r] / lrun[r];
    }
}

// ---------------------------------------------------------------------------
// Host launcher
// ---------------------------------------------------------------------------
extern "C" void kernel_launch(void* const* d_in, const int* in_sizes, int n_in,
                              void* d_out, int out_size, void* d_ws, size_t ws_size,
                              hipStream_t stream) {
  (void)in_sizes; (void)n_in; (void)out_size; (void)ws_size;
  const float* X  = (const float*)d_in[0];
  const float* Wq = (const float*)d_in[1];
  const float* bq = (const float*)d_in[2];
  const float* Wk = (const float*)d_in[3];
  const float* bk = (const float*)d_in[4];
  const float* Wv = (const float*)d_in[5];
  const float* bv = (const float*)d_in[6];
  float* Out = (float*)d_out;

  // Workspace layout (f16 elements): X(8M) Wq Wk Wv (1M ea) Q K V (8M ea) ~70MB.
  _Float16* ws  = (_Float16*)d_ws;
  _Float16* Xh  = ws;
  _Float16* Wqh = Xh  + 8388608;
  _Float16* Wkh = Wqh + 1048576;
  _Float16* Wvh = Wkh + 1048576;
  _Float16* Qh  = Wvh + 1048576;
  _Float16* Kh  = Qh  + 8388608;
  _Float16* Vh  = Kh  + 8388608;

  cvt_f32_f16<<<8192, 256, 0, stream>>>(X,  Xh,  8388608 / 4);
  cvt_f32_f16<<<1024, 256, 0, stream>>>(Wq, Wqh, 1048576 / 4);
  cvt_f32_f16<<<1024, 256, 0, stream>>>(Wk, Wkh, 1048576 / 4);
  cvt_f32_f16<<<1024, 256, 0, stream>>>(Wv, Wvh, 1048576 / 4);

  dim3 g1(64, 16);
  qkv_gemm<<<g1, 256, 0, stream>>>(Xh, Wqh, bq, Qh);
  qkv_gemm<<<g1, 256, 0, stream>>>(Xh, Wkh, bk, Kh);
  qkv_gemm<<<g1, 256, 0, stream>>>(Xh, Wvh, bv, Vh);

  dim3 g2(16, 16, 4);
  attention<<<g2, 256, 0, stream>>>(Qh, Kh, Vh, Out);
}
